// SCAEAttention_9311489097974
// MI455X (gfx1250) — compile-verified, matrix operand order checked by reading, not run on hardware
//
#include <hip/hip_runtime.h>
#include <hip/hip_bf16.h>
#include <stdint.h>

// ---------------------------------------------------------------------------
// Problem dims (fixed by the reference)
// ---------------------------------------------------------------------------
#define BB 4
#define SS 256
#define HH 12
#define DHH 64
#define DD 768
#define FDD 2048
#define FUU 2048
#define BS (BB * SS)   // 1024

typedef __attribute__((ext_vector_type(16))) _Float16 v16h;
typedef __attribute__((ext_vector_type(8)))  float    v8f;

union Frag {
    v16h v;
    uint32_t u[8];
};

// Wait on ASYNCcnt with a compile-time immediate.
template <int N>
__device__ __forceinline__ void wait_async()
{
    if constexpr (N == 0)
        asm volatile("s_wait_asynccnt 0x0" ::: "memory");
    else
        asm volatile("s_wait_asynccnt 0x4" ::: "memory");
}

// Async DMA 16 bytes global -> LDS (per lane), tracked by ASYNCcnt.
__device__ __forceinline__ void async_b128(uint32_t lds_byte_off, const void* gptr)
{
    asm volatile("global_load_async_to_lds_b128 %0, %1, off"
                 :: "v"(lds_byte_off), "v"((uint64_t)(uintptr_t)gptr)
                 : "memory");
}

// ---------------------------------------------------------------------------
// Tiled WMMA GEMM (all-NT):  C[M,N] (+)= A[M,K] . B[N,K]^T
//   A row-major [M,K], B row-major [N,K]  (both f16)
//   OUT_F32 : C is float, else _Float16
//   ACCUM   : C += result (RMW), else C = result
//   MASK    : zero C element (row,col) where mask[row*ldm+col]==0
//   OUT_T   : store transposed: C[col*ldc + row]
// Block: 256 threads = 8 waves -> 128x128 C tile; wave = 32(M) x 64(N),
// i.e. 2 A-frags, 4 B-frags, 8 v_wmma_f32_16x16x32_f16 per 32-wide K step.
// Global->LDS staging uses GLOBAL_LOAD_ASYNC_TO_LDS_B128, double buffered.
// Requires M%128==0, N%128==0, K%32==0 (true for every call below).
// ---------------------------------------------------------------------------
template <bool OUT_F32, bool ACCUM, bool MASK, bool OUT_T>
__global__ __launch_bounds__(256)
void gemm_wmma(const _Float16* __restrict__ A, int lda,
               const _Float16* __restrict__ Bm, int ldb,
               void* __restrict__ Cv, int ldc,
               const int* __restrict__ mask, int ldm,
               int K)
{
    __shared__ __align__(16) _Float16 As[2][128 * 32];
    __shared__ __align__(16) _Float16 Bs[2][128 * 32];

    const int tid   = threadIdx.x;
    const int lane  = tid & 31;
    const int wave  = tid >> 5;          // 0..7
    const int wm    = wave & 3;          // 32-row slice within 128
    const int wn    = wave >> 2;         // 64-col slice within 128
    const int mBase = blockIdx.y * 128;
    const int nBase = blockIdx.x * 128;
    const int klo   = (lane >> 4) * 8;   // K base for this lane half
    const int row16 = lane & 15;

    // Issue the async loads for one 128x32 A tile + 128x32 B tile.
    // 512 16B chunks per tile / 256 threads = 2 chunks each => 4 async ops
    // per thread (A+B) per K step.
    auto stage = [&](int k0, int buf) {
        #pragma unroll
        for (int it = 0; it < 2; ++it) {
            const int q = it * 256 + tid;    // 0..511
            const int r = q >> 2;            // row 0..127
            const int c = (q & 3) * 8;       // col in f16 elems (16B chunks)
            async_b128((uint32_t)(uintptr_t)&As[buf][r * 32 + c],
                       A + (size_t)(mBase + r) * lda + k0 + c);
            async_b128((uint32_t)(uintptr_t)&Bs[buf][r * 32 + c],
                       Bm + (size_t)(nBase + r) * ldb + k0 + c);
        }
    };

    v8f acc[2][4] = {};

    stage(0, 0);
    int buf = 0;
    for (int k0 = 0; k0 < K; k0 += 32) {
        // All waves done computing on buf^1 before it gets overwritten.
        __syncthreads();
        if (k0 + 32 < K) {
            stage(k0 + 32, buf ^ 1);
            wait_async<4>();   // current tile's 4 DMAs done; next 4 in flight
        } else {
            wait_async<0>();
        }
        __syncthreads();       // every wave's current-tile DMAs complete

        // A fragments (ISA 16-bit A-matrix 16x32 layout) -> ds_load_b128 pairs
        Frag fa[2];
        #pragma unroll
        for (int at = 0; at < 2; ++at) {
            const int arow = wm * 32 + at * 16 + row16;
            #pragma unroll
            for (int i = 0; i < 8; ++i) {
                const int kk = klo + 2 * i + ((i >= 4) ? 8 : 0);
                fa[at].u[i] = *(const uint32_t*)&As[buf][arow * 32 + kk];
            }
        }
        #pragma unroll
        for (int nt = 0; nt < 4; ++nt) {
            Frag fb;
            const int bcol = wn * 64 + nt * 16 + row16;
            #pragma unroll
            for (int i = 0; i < 8; ++i) {
                const int kk = klo + 2 * i + ((i >= 4) ? 8 : 0);
                fb.u[i] = *(const uint32_t*)&Bs[buf][bcol * 32 + kk];
            }
            #pragma unroll
            for (int at = 0; at < 2; ++at) {
                acc[at][nt] = __builtin_amdgcn_wmma_f32_16x16x32_f16(
                    false, fa[at].v, false, fb.v, (short)0, acc[at][nt],
                    false, false);
            }
        }
        buf ^= 1;
    }

    // ---- store: C/D 16x16 f32 layout (VGPR r -> M=r / r+8 by lane half) ----
    const int crow0 = mBase + wm * 32 + (lane >> 4) * 8;
    const int ccol0 = nBase + wn * 64 + (lane & 15);
    #pragma unroll
    for (int at = 0; at < 2; ++at) {
        #pragma unroll
        for (int nt = 0; nt < 4; ++nt) {
            const int ccol = ccol0 + nt * 16;
            #pragma unroll
            for (int r = 0; r < 8; ++r) {
                const int crow = crow0 + at * 16 + r;
                float val = acc[at][nt][r];
                if (MASK) {
                    if (mask[(size_t)crow * ldm + ccol] == 0) val = 0.0f;
                }
                const size_t idx = OUT_T ? ((size_t)ccol * ldc + crow)
                                         : ((size_t)crow * ldc + ccol);
                if (OUT_F32) {
                    float* C = (float*)Cv;
                    C[idx] = ACCUM ? (C[idx] + val) : val;
                } else {
                    _Float16* C = (_Float16*)Cv;
                    if (ACCUM) val += (float)C[idx];
                    C[idx] = (_Float16)val;
                }
            }
        }
    }
}

// ---------------------------------------------------------------------------
// Small elementwise / conversion kernels
// ---------------------------------------------------------------------------
__global__ void k_cvt(const float* __restrict__ in, _Float16* __restrict__ out,
                      long long n)
{
    long long i  = (long long)blockIdx.x * blockDim.x + threadIdx.x;
    long long st = (long long)gridDim.x * blockDim.x;
    for (; i < n; i += st) out[i] = (_Float16)in[i];
}

// in: [batch][R][C] f32  ->  out: [batch][C][R] f16
__global__ void k_transpose_cvt(const float* __restrict__ in,
                                _Float16* __restrict__ out,
                                int batch, int R, int C)
{
    long long n  = (long long)batch * R * C;
    long long i  = (long long)blockIdx.x * blockDim.x + threadIdx.x;
    long long st = (long long)gridDim.x * blockDim.x;
    for (; i < n; i += st) {
        long long rc  = (long long)R * C;
        int b = (int)(i / rc);
        long long rem = i - (long long)b * rc;
        int r = (int)(rem / C);
        int c = (int)(rem - (long long)r * C);
        out[(long long)b * rc + (long long)c * R + r] = (_Float16)in[i];
    }
}

// out[i] = x[i] / ln[i / cols]
__global__ void k_lndiv(const float* __restrict__ x, const float* __restrict__ ln,
                        _Float16* __restrict__ out, long long rows, int cols)
{
    long long n  = rows * cols;
    long long i  = (long long)blockIdx.x * blockDim.x + threadIdx.x;
    long long st = (long long)gridDim.x * blockDim.x;
    for (; i < n; i += st) {
        long long r = i / cols;
        out[i] = (_Float16)(x[i] / ln[r]);
    }
}

// w2[d] = sum_h sum_m W_OVt[h][d][m] * up_b_dec[m]
__global__ void k_w2(const _Float16* __restrict__ WOVt,
                     const float* __restrict__ ubd, float* __restrict__ w2)
{
    int d = blockIdx.x * blockDim.x + threadIdx.x;
    if (d >= DD) return;
    float s = 0.0f;
    for (int h = 0; h < HH; ++h) {
        const _Float16* row = WOVt + ((long long)h * DD + d) * DD;
        for (int m = 0; m < DD; ++m) s += (float)row[m] * ubd[m];
    }
    w2[d] = s;
}

// c0[f] = enc_b[f] - enc_W[f,:].b_dec ;  c1[f] = enc_W[f,:].w2
__global__ void k_c0c1(const float* __restrict__ encW,
                       const float* __restrict__ encb,
                       const float* __restrict__ bdec,
                       const float* __restrict__ w2,
                       float* __restrict__ c0, float* __restrict__ c1)
{
    int f = blockIdx.x * blockDim.x + threadIdx.x;
    if (f >= FDD) return;
    const float* row = encW + (long long)f * DD;
    float s0 = 0.0f, s1 = 0.0f;
    for (int d = 0; d < DD; ++d) {
        float w = row[d];
        s0 += w * bdec[d];
        s1 += w * w2[d];
    }
    c0[f] = encb[f] - s0;
    c1[f] = s1;
}

// out[r][f] = c0[f] + c1[f] / ln[r]
__global__ void k_init_out(const float* __restrict__ ln,
                           const float* __restrict__ c0,
                           const float* __restrict__ c1,
                           float* __restrict__ out)
{
    long long n  = (long long)BS * FDD;
    long long i  = (long long)blockIdx.x * blockDim.x + threadIdx.x;
    long long st = (long long)gridDim.x * blockDim.x;
    for (; i < n; i += st) {
        int r = (int)(i >> 11);
        int f = (int)(i & 2047);
        out[i] = c0[f] + c1[f] / ln[r];
    }
}

// ---------------------------------------------------------------------------
// Host side
// ---------------------------------------------------------------------------
extern "C" void kernel_launch(void* const* d_in, const int* in_sizes, int n_in,
                              void* d_out, int out_size, void* d_ws, size_t ws_size,
                              hipStream_t stream)
{
    (void)in_sizes; (void)n_in; (void)out_size; (void)ws_size;

    const float* resid   = (const float*)d_in[0];   // [4,256,768]
    const float* lns     = (const float*)d_in[1];   // [4,256,1] -> flat [1024]
    const float* probs   = (const float*)d_in[2];   // [4,12,256,256]
    const float* W_O     = (const float*)d_in[3];   // [12,64,768]
    const float* W_V     = (const float*)d_in[4];   // [12,768,64]
    const float* enc_W   = (const float*)d_in[5];   // [2048,768]
    const float* enc_b   = (const float*)d_in[6];   // [2048]
    const float* b_dec   = (const float*)d_in[7];   // [768]
    const float* updW    = (const float*)d_in[8];   // [768,2048]
    const float* up_bdec = (const float*)d_in[9];   // [768]
    const float* pruned  = (const float*)d_in[10];  // [4,256,2048]
    const int*   cmask   = (const int*)d_in[11];    // [2048,2048]

    float* out = (float*)d_out;                     // [4,256,2048] fp32

    // ---- workspace carve-out (256B aligned slabs) ----
    size_t off = 0;
    auto carve = [&](size_t elems, size_t esz) -> void* {
        void* p = (char*)d_ws + off;
        off += (elems * esz + 255) & ~(size_t)255;
        return p;
    };
    _Float16* encW_h  = (_Float16*)carve((size_t)FDD * DD, 2);       // [f][o]
    _Float16* WOt_h   = (_Float16*)carve((size_t)HH * DD * DHH, 2);  // [h][o][k]
    _Float16* WV_h    = (_Float16*)carve((size_t)HH * DD * DHH, 2);  // [h][m][k]
    _Float16* WOV_h   = (_Float16*)carve((size_t)HH * DD * DD, 2);   // [h][m][o]
    _Float16* WOVt_h  = (_Float16*)carve((size_t)HH * DD * DD, 2);   // [h][o][m]
    _Float16* actln_h = (_Float16*)carve((size_t)BS * DD, 2);        // [bs][m]
    _Float16* upln_h  = (_Float16*)carve((size_t)BS * FUU, 2);       // [bk][u]
    _Float16* updWt_h = (_Float16*)carve((size_t)FUU * DD, 2);       // [u][i]
    _Float16* t_h     = (_Float16*)carve((size_t)HH * BS * DD, 2);   // [h][bs][o]
    _Float16* pmt_h   = (_Float16*)carve((size_t)HH * FDD * BS, 2);  // [h][f][bs]
    _Float16* A1_h    = (_Float16*)carve((size_t)HH * FDD * DD, 2);  // [h][f][i]
    _Float16* virt_h  = (_Float16*)carve((size_t)HH * FDD * FUU, 2); // [h][f][u]
    _Float16* probs_h = (_Float16*)carve((size_t)BB * HH * SS * SS, 2);
    float*    w2_f    = (float*)carve(DD, 4);
    float*    c0_f    = (float*)carve(FDD, 4);
    float*    c1_f    = (float*)carve(FDD, 4);

    const dim3 blk(256);
    const dim3 g4k(4096);

    // ---- conversions ----
    k_cvt<<<g4k, blk, 0, stream>>>(enc_W, encW_h, (long long)FDD * DD);
    k_cvt<<<g4k, blk, 0, stream>>>(W_V, WV_h, (long long)HH * DD * DHH);
    k_cvt<<<g4k, blk, 0, stream>>>(probs, probs_h, (long long)BB * HH * SS * SS);
    k_transpose_cvt<<<g4k, blk, 0, stream>>>(W_O, WOt_h, HH, DHH, DD);
    k_transpose_cvt<<<g4k, blk, 0, stream>>>(updW, updWt_h, 1, DD, FUU);
    k_lndiv<<<g4k, blk, 0, stream>>>(resid, lns, actln_h, BS, DD);
    k_lndiv<<<g4k, blk, 0, stream>>>(pruned, lns, upln_h, BS, FUU);

    // ---- per-head weight products: W_OV[h][m][o], W_OVt[h][o][m] ----
    for (int h = 0; h < HH; ++h) {
        dim3 g(DD / 128, DD / 128);
        gemm_wmma<false, false, false, false><<<g, 256, 0, stream>>>(
            WV_h + (size_t)h * DD * DHH, DHH,
            WOt_h + (size_t)h * DD * DHH, DHH,
            WOV_h + (size_t)h * DD * DD, DD, nullptr, 0, DHH);
        gemm_wmma<false, false, false, false><<<g, 256, 0, stream>>>(
            WOt_h + (size_t)h * DD * DHH, DHH,
            WV_h + (size_t)h * DD * DHH, DHH,
            WOVt_h + (size_t)h * DD * DD, DD, nullptr, 0, DHH);
    }

    // ---- bias terms -> initialize d_out ----
    k_w2<<<dim3(3), blk, 0, stream>>>(WOVt_h, up_bdec, w2_f);
    k_c0c1<<<dim3(8), blk, 0, stream>>>(enc_W, enc_b, b_dec, w2_f, c0_f, c1_f);
    k_init_out<<<dim3(2048), blk, 0, stream>>>(lns, c0_f, c1_f, out);

    // ---- t[h] = act_ln @ W_OV[h] (dot rows of W_OVt) ----
    for (int h = 0; h < HH; ++h) {
        dim3 g(DD / 128, BS / 128);
        gemm_wmma<false, false, false, false><<<g, 256, 0, stream>>>(
            actln_h, DD,
            WOVt_h + (size_t)h * DD * DD, DD,
            t_h + (size_t)h * BS * DD, DD, nullptr, 0, DD);
    }
    // ---- pm_t[h][f][bs] = (t[h] @ enc_W^T)^T  (transposed store) ----
    for (int h = 0; h < HH; ++h) {
        dim3 g(FDD / 128, BS / 128);
        gemm_wmma<false, false, false, true><<<g, 256, 0, stream>>>(
            t_h + (size_t)h * BS * DD, DD,
            encW_h, DD,
            pmt_h + (size_t)h * FDD * BS, BS, nullptr, 0, DD);
    }
    // ---- A1[h] = enc_W @ W_OV[h]^T ----
    for (int h = 0; h < HH; ++h) {
        dim3 g(DD / 128, FDD / 128);
        gemm_wmma<false, false, false, false><<<g, 256, 0, stream>>>(
            encW_h, DD,
            WOV_h + (size_t)h * DD * DD, DD,
            A1_h + (size_t)h * FDD * DD, DD, nullptr, 0, DD);
    }
    // ---- virtual[h] = (A1[h] @ up_dec_W) * mask ----
    for (int h = 0; h < HH; ++h) {
        dim3 g(FUU / 128, FDD / 128);
        gemm_wmma<false, false, true, false><<<g, 256, 0, stream>>>(
            A1_h + (size_t)h * FDD * DD, DD,
            updWt_h, DD,
            virt_h + (size_t)h * FDD * FUU, FUU, cmask, FUU, DD);
    }
    // ---- pm_t[h] += (up_ln @ virtual[h]^T)^T  (transposed RMW) ----
    for (int h = 0; h < HH; ++h) {
        dim3 g(FDD / 128, BS / 128);
        gemm_wmma<false, true, false, true><<<g, 256, 0, stream>>>(
            upln_h, FUU,
            virt_h + (size_t)h * FDD * FUU, FUU,
            pmt_h + (size_t)h * FDD * BS, BS, nullptr, 0, FUU);
    }
    // ---- out[b] += sum_h probs[b,h] @ pm[h][b] : NT vs pm_t rows ----
    for (int b = 0; b < BB; ++b) {
        for (int h = 0; h < HH; ++h) {
            dim3 g(FDD / 128, SS / 128);
            gemm_wmma<true, true, false, false><<<g, 256, 0, stream>>>(
                probs_h + ((size_t)b * HH + h) * SS * SS, SS,
                pmt_h + (size_t)h * FDD * BS + (size_t)b * SS, BS,
                out + (size_t)b * SS * FDD, FDD, nullptr, 0, SS);
        }
    }
}
